// EquiformerV3_OC_42305427866219
// MI455X (gfx1250) — compile-verified, matrix operand order checked by reading, not run on hardware
//
#include <hip/hip_runtime.h>
#include <hip/hip_fp16.h>

typedef __attribute__((ext_vector_type(16))) _Float16 v16h;
typedef __attribute__((ext_vector_type(8)))  _Float16 v8h;
typedef __attribute__((ext_vector_type(8)))  float    v8f;

#define C_      128
#define KM_     29
#define KF_     49
#define H_      8
#define ALPHA_  32
#define VAL_    16
#define HID_    64
#define FFN_    128
#define NL_     4
#define NRBF_   600
#define CUTOFF_ 12.0f
#define AVG_DEG_   23.395238876342773f
#define AVG_NODES_ 77.81317f
#define XROW_ (KF_*C_)    /* 6272 */
#define MROW_ (KM_*HID_)  /* 1856 */

__constant__ int c_M2F[KM_] = {0, 1,2,3, 4,5,6,7,8, 10,11,12,13,14,
                               18,19,20,21,22, 28,29,30,31,32, 40,41,42,43,44};
__constant__ int c_LOF[KM_] = {0, 1,1,1, 2,2,2,2,2, 3,3,3,3,3,
                               4,4,4,4,4, 5,5,5,5,5, 6,6,6,6,6};

__device__ __forceinline__ float siluf(float x) {
  return x * (1.0f / (1.0f + __expf(-x)));
}

__device__ __forceinline__ v8f wmma32(v16h a, v16h b, v8f c) {
  return __builtin_amdgcn_wmma_f32_16x16x32_f16(false, a, false, b,
                                                (short)0, c, false, false);
}

// A fragment from LDS: row-major [16][lda] halves, K-tile at k0.
// Lane layout (16-bit A 16x32): m = lane&15; halves 0..7 -> K = 8*hi + 0..7,
// halves 8..15 -> K = 16 + 8*hi + 0..7.
__device__ __forceinline__ v16h fragA(const _Float16* lds, int lda, int k0) {
  const int lane = threadIdx.x & 31;
  const int m = lane & 15, hi = lane >> 4;
  const _Float16* p = lds + m * lda + k0 + 8 * hi;
  v8h lo = *(const v8h*)p;
  v8h hh = *(const v8h*)(p + 16);
  v16h r;
#pragma unroll
  for (int j = 0; j < 8; ++j) { r[j] = lo[j]; r[j + 8] = hh[j]; }
  return r;
}

// B fragment from LDS: column-major [ncols][ldb] halves (lds[n*ldb+k]).
__device__ __forceinline__ v16h fragB(const _Float16* lds, int ldb, int n0, int k0) {
  const int lane = threadIdx.x & 31;
  const int n = lane & 15, hi = lane >> 4;
  const _Float16* p = lds + (size_t)(n0 + n) * ldb + k0 + 8 * hi;
  v8h lo = *(const v8h*)p;
  v8h hh = *(const v8h*)(p + 16);
  v16h r;
#pragma unroll
  for (int j = 0; j < 8; ++j) { r[j] = lo[j]; r[j + 8] = hh[j]; }
  return r;
}

// float <-> monotonic uint key (for float atomicMax)
__device__ __forceinline__ unsigned f2key(float f) {
  unsigned u = __float_as_uint(f);
  return (u & 0x80000000u) ? ~u : (u | 0x80000000u);
}
__device__ __forceinline__ float key2f(unsigned k) {
  unsigned u = (k & 0x80000000u) ? (k & 0x7fffffffu) : ~k;
  return __uint_as_float(u);
}

// ---------------------------------------------------------------- utilities
__global__ void k_fill(float* p, long n, float v) {
  long i = (long)blockIdx.x * blockDim.x + threadIdx.x;
  long st = (long)gridDim.x * blockDim.x;
  for (; i < n; i += st) p[i] = v;
}

__global__ void k_init_x(float* x, const float* emb, const int* an, int N) {
  int n = blockIdx.x;
  if (n >= N) return;
  float* row = x + (long)n * XROW_;
  for (int i = threadIdx.x; i < XROW_; i += blockDim.x) row[i] = 0.f;
  __syncthreads();
  int a = an[n];
  for (int c = threadIdx.x; c < C_; c += blockDim.x) row[c] = emb[(long)a * C_ + c];
}

// --------------------------------------------- GEMM with on-the-fly RBF A
// out[e][128] = silu(rbf(d) @ W + bias); 64 edges x 64 cols per block.
__global__ __launch_bounds__(128)
void k_rbf_gemm(const float* dist, const float* W, const float* bias,
                _Float16* out, int E) {
  __shared__ __align__(16) _Float16 sA[64 * 32];
  __shared__ __align__(16) _Float16 sB[64 * 32];
  __shared__ float sD[64];
  const int e0 = blockIdx.x * 64;
  const int n0 = blockIdx.y * 64;
  const int tid = threadIdx.x, w = tid >> 5;
  if (tid < 64) sD[tid] = (e0 + tid < E) ? dist[e0 + tid] : 0.f;
  const float step = CUTOFF_ / (NRBF_ - 1);
  const float coeff = -0.5f / ((2.f * step) * (2.f * step));
  const v8f vz = {};
  v8f acc[4] = {vz, vz, vz, vz};
  __syncthreads();
  for (int ks = 0; ks < 19; ++ks) {
    __syncthreads();
    for (int idx = tid; idx < 2048; idx += 128) {
      int m = idx >> 5, k = idx & 31;
      int kk = ks * 32 + k;
      float v = 0.f;
      if (kk < NRBF_) { float dd = sD[m] - (float)kk * step; v = __expf(coeff * dd * dd); }
      sA[m * 32 + k] = (_Float16)v;
    }
    for (int idx = tid; idx < 2048; idx += 128) {
      int n = idx >> 5, k = idx & 31;
      int kk = ks * 32 + k;
      sB[n * 32 + k] = (_Float16)((kk < NRBF_) ? W[(long)kk * C_ + n0 + n] : 0.f);
    }
    if (ks + 1 < 19 && (ks + 1) * 32 + (tid >> 2) < NRBF_)
      __builtin_prefetch((const void*)(W + (long)((ks + 1) * 32 + (tid >> 2)) * C_ +
                                      n0 + ((tid & 3) << 4)), 0, 1);
    __syncthreads();
    v16h bf = fragB(sB, 32, w * 16, 0);
#pragma unroll
    for (int rt = 0; rt < 4; ++rt)
      acc[rt] = wmma32(fragA(sA + rt * 16 * 32, 32, 0), bf, acc[rt]);
  }
  const int lane = tid & 31, nn = lane & 15, hi = lane >> 4;
  const int col = n0 + w * 16 + nn;
  float b = bias ? bias[col] : 0.f;
#pragma unroll
  for (int rt = 0; rt < 4; ++rt) {
#pragma unroll
    for (int r = 0; r < 8; ++r) {
      int e = e0 + rt * 16 + r + 8 * hi;
      if (e < E) out[(long)e * C_ + col] = (_Float16)siluf(acc[rt][r] + b);
    }
  }
}

// ----------------------------- generic GEMM: out_f16 = silu(A@B + bias)
// A: f16 or f32, row stride lda, K in {64,128}; B f32 [K][ldb].
// 64 rows x 64 cols per block (4 row-tiles per wave share one B fragment).
__global__ __launch_bounds__(128)
void k_gemm_silu(const void* A, int a_f32, int lda, const float* B, int ldb,
                 const float* bias, _Float16* out, int ldo, int M, int K) {
  __shared__ __align__(16) _Float16 sA[64 * 128];
  __shared__ __align__(16) _Float16 sB[64 * 32];
  const int r0 = blockIdx.x * 64;
  const int n0 = blockIdx.y * 64;
  const int tid = threadIdx.x, w = tid >> 5;
  if (a_f32) {
    const float* Af = (const float*)A;
    for (int idx = tid; idx < 64 * K; idx += 128) {
      int r = idx / K, k = idx - r * K;
      sA[idx] = (_Float16)((r0 + r < M) ? Af[(long)(r0 + r) * lda + k] : 0.f);
    }
  } else {
    const _Float16* Ah = (const _Float16*)A;
    int cpr = K >> 3;
    for (int idx = tid; idx < 64 * cpr; idx += 128) {
      int r = idx / cpr, cc = (idx - r * cpr) * 8;
      v8h v = {};
      if (r0 + r < M) v = *(const v8h*)(Ah + (long)(r0 + r) * lda + cc);
      *(v8h*)(sA + r * K + cc) = v;
    }
  }
  const v8f vz = {};
  v8f acc[4] = {vz, vz, vz, vz};
  const int ksteps = K >> 5;
  for (int ks = 0; ks < ksteps; ++ks) {
    __syncthreads();
    for (int idx = tid; idx < 2048; idx += 128) {
      int n = idx >> 5, k = idx & 31;
      sB[n * 32 + k] = (_Float16)B[(long)(ks * 32 + k) * ldb + n0 + n];
    }
    if (ks + 1 < ksteps)
      __builtin_prefetch((const void*)(B + (long)((ks + 1) * 32 + (tid >> 2)) * ldb +
                                      n0 + ((tid & 3) << 4)), 0, 1);
    __syncthreads();
    v16h bf = fragB(sB, 32, w * 16, 0);
#pragma unroll
    for (int rt = 0; rt < 4; ++rt)
      acc[rt] = wmma32(fragA(sA + rt * 16 * K, K, ks * 32), bf, acc[rt]);
  }
  const int lane = tid & 31, nn = lane & 15, hi = lane >> 4;
  const int col = n0 + w * 16 + nn;
  float b = bias ? bias[col] : 0.f;
#pragma unroll
  for (int rt = 0; rt < 4; ++rt) {
#pragma unroll
    for (int r = 0; r < 8; ++r) {
      int row = r0 + rt * 16 + r + 8 * hi;
      if (row < M) out[(long)row * ldo + col] = (_Float16)siluf(acc[rt][r] + b);
    }
  }
}

// ------------------- degree projection + scatter: x[tgt, M2F, :] += h@Wp /deg
// 64 edges x 64 (of 3712) cols per block.
__global__ __launch_bounds__(128)
void k_deg_scatter(const _Float16* hE, const float* Wp, const int* tgt,
                   float* x, int E) {
  __shared__ __align__(16) _Float16 sA[64 * 128];
  __shared__ __align__(16) _Float16 sB[64 * 32];
  __shared__ int sT[64];
  const int e0 = blockIdx.x * 64;
  const int n0 = blockIdx.y * 64;
  const int tid = threadIdx.x, w = tid >> 5;
  if (tid < 64) sT[tid] = (e0 + tid < E) ? tgt[e0 + tid] : 0;
  for (int idx = tid; idx < 1024; idx += 128) {
    int r = idx >> 4, cc = (idx & 15) * 8;
    v8h v = {};
    if (e0 + r < E) v = *(const v8h*)(hE + (long)(e0 + r) * C_ + cc);
    *(v8h*)(sA + r * C_ + cc) = v;
  }
  const v8f vz = {};
  v8f acc[4] = {vz, vz, vz, vz};
  for (int ks = 0; ks < 4; ++ks) {
    __syncthreads();
    for (int idx = tid; idx < 2048; idx += 128) {
      int n = idx >> 5, k = idx & 31;
      sB[n * 32 + k] = (_Float16)Wp[(long)(ks * 32 + k) * (KM_ * C_) + n0 + n];
    }
    if (ks + 1 < 4)
      __builtin_prefetch((const void*)(Wp + (long)((ks + 1) * 32 + (tid >> 2)) *
                                      (KM_ * C_) + n0 + ((tid & 3) << 4)), 0, 1);
    __syncthreads();
    v16h bf = fragB(sB, 32, w * 16, 0);
#pragma unroll
    for (int rt = 0; rt < 4; ++rt)
      acc[rt] = wmma32(fragA(sA + rt * 16 * 128, 128, ks * 32), bf, acc[rt]);
  }
  const int lane = tid & 31, nn = lane & 15, hi = lane >> 4;
  const int gcol = n0 + w * 16 + nn;
  const int f = c_M2F[gcol >> 7], c = gcol & 127;
  const float sc = 1.0f / AVG_DEG_;
#pragma unroll
  for (int rt = 0; rt < 4; ++rt) {
#pragma unroll
    for (int r = 0; r < 8; ++r) {
      int m = rt * 16 + r + 8 * hi, e = e0 + m;
      if (e < E) atomicAdd(&x[(long)sT[m] * XROW_ + f * C_ + c], acc[rt][r] * sc);
    }
  }
}

// --------------------------------- RMS norm rows of x -> f16 [N][nk][128]
__global__ void k_rms(const float* x, const float* gamma, _Float16* out,
                      int N, int nk, int use_m2f) {
  __shared__ float sW[4];
  long b = blockIdx.x;
  int n = (int)(b / nk), kk = (int)(b % nk);
  if (n >= N) return;
  int f = use_m2f ? c_M2F[kk] : kk;
  const float* row = x + (long)n * XROW_ + f * C_;
  int tid = threadIdx.x;
  float v = row[tid];
  float ss = v * v;
  for (int off = 16; off; off >>= 1) ss += __shfl_down(ss, off, 32);
  if ((tid & 31) == 0) sW[tid >> 5] = ss;
  __syncthreads();
  float tot = sW[0] + sW[1] + sW[2] + sW[3];
  float sc = rsqrtf(tot * (1.0f / C_) + 1e-6f);
  out[b * C_ + tid] = (_Float16)(v * sc * gamma[tid]);
}

// --------------------------------- radl[e][l] = rad[e]·Wrl[:,l]
__global__ void k_radl(const _Float16* rad, const float* Wrl, float* radl, int E) {
  int e = blockIdx.x * blockDim.x + threadIdx.x;
  if (e >= E) return;
  float acc[7] = {0, 0, 0, 0, 0, 0, 0};
  const _Float16* r = rad + (long)e * C_;
  for (int c = 0; c < C_; ++c) {
    float rv = (float)r[c];
#pragma unroll
    for (int l = 0; l < 7; ++l) acc[l] += rv * Wrl[c * 7 + l];
  }
#pragma unroll
  for (int l = 0; l < 7; ++l) radl[(long)e * 7 + l] = acc[l];
}

// -------- message GEMM: m[e][k][:] = (ym[src]||ym[tgt]) @ Wmsg * radl[LOF]
__global__ __launch_bounds__(128)
void k_msg(const _Float16* ym, const int* src, const int* tgt,
           const float* Wmsg, const float* radl, _Float16* m_out, int E) {
  __shared__ __align__(16) _Float16 sB[64 * 256];  // [n][k], resident all 29 reuses
  __shared__ __align__(16) _Float16 sA[16 * 256];
  __shared__ int sS[16], sT[16];
  const int e0 = blockIdx.x * 16;
  const int tid = threadIdx.x, w = tid >> 5;
  if (tid < 16) {
    int e = e0 + tid;
    sS[tid] = (e < E) ? src[e] : 0;
    sT[tid] = (e < E) ? tgt[e] : 0;
  }
  for (int idx = tid; idx < 64 * 256; idx += 128) {
    int n = idx >> 8, k = idx & 255;
    sB[idx] = (_Float16)Wmsg[(long)k * HID_ + n];
  }
  __syncthreads();
  const int lane = tid & 31, nn = lane & 15, hi = lane >> 4;
  const int col = w * 16 + nn;
  for (int kk = 0; kk < KM_; ++kk) {
    __syncthreads();
    for (int idx = tid; idx < 512; idx += 128) {
      int r = idx >> 5, cc = (idx & 31) * 8;
      int node = (cc < 128) ? sS[r] : sT[r];
      int c = cc & 127;
      v8h v = *(const v8h*)(ym + (long)node * (KM_ * C_) + kk * C_ + c);
      *(v8h*)(sA + r * 256 + cc) = v;
    }
    __syncthreads();
    v8f acc = {};
#pragma unroll
    for (int ks = 0; ks < 8; ++ks)
      acc = wmma32(fragA(sA, 256, ks * 32), fragB(sB, 256, w * 16, ks * 32), acc);
    int lof = c_LOF[kk];
#pragma unroll
    for (int r = 0; r < 8; ++r) {
      int mrow = r + 8 * hi, e = e0 + mrow;
      if (e < E) {
        float scl = radl[(long)e * 7 + lof];
        m_out[(long)e * MROW_ + kk * HID_ + col] = (_Float16)(acc[r] * scl);
      }
    }
  }
}

// ------- attention projection: logits[e][h] += silu(m0@Wap)[h,a]*wa[h,a]
__global__ __launch_bounds__(128)
void k_ap(const _Float16* m, const float* Wap, const float* wa,
          float* logits, int E) {
  __shared__ __align__(16) _Float16 sA[16 * 64];
  __shared__ __align__(16) _Float16 sB[64 * 32];
  const int e0 = blockIdx.x * 16;
  const int n0 = blockIdx.y * 64;
  const int tid = threadIdx.x, w = tid >> 5;
  for (int idx = tid; idx < 128; idx += 128) {
    int r = idx >> 3, cc = (idx & 7) * 8;
    v8h v = {};
    if (e0 + r < E) v = *(const v8h*)(m + (long)(e0 + r) * MROW_ + cc);
    *(v8h*)(sA + r * 64 + cc) = v;
  }
  v8f acc = {};
  for (int ks = 0; ks < 2; ++ks) {
    __syncthreads();
    for (int idx = tid; idx < 2048; idx += 128) {
      int n = idx >> 5, k = idx & 31;
      sB[n * 32 + k] = (_Float16)Wap[(long)(ks * 32 + k) * (H_ * ALPHA_) + n0 + n];
    }
    __syncthreads();
    acc = wmma32(fragA(sA, 64, ks * 32), fragB(sB, 32, w * 16, 0), acc);
  }
  const int lane = tid & 31, nn = lane & 15, hi = lane >> 4;
  const int j = n0 + w * 16 + nn;
  const int h = j >> 5, al = j & 31;
  const float wv = wa[h * ALPHA_ + al];
#pragma unroll
  for (int r = 0; r < 8; ++r) {
    int e = e0 + r + 8 * hi;
    if (e < E) atomicAdd(&logits[(long)e * H_ + h], siluf(acc[r]) * wv);
  }
}

__global__ void k_segmax(const float* logits, const int* tgt, unsigned* smax, int E) {
  long i = (long)blockIdx.x * blockDim.x + threadIdx.x;
  if (i >= (long)E * H_) return;
  int e = (int)(i >> 3), h = (int)(i & 7);
  atomicMax(&smax[(long)tgt[e] * H_ + h], f2key(logits[i]));
}

__global__ void k_exp_env(const float* logits, const unsigned* smax, const float* dist,
                          const int* tgt, float* ex, float* den, int E) {
  long i = (long)blockIdx.x * blockDim.x + threadIdx.x;
  if (i >= (long)E * H_) return;
  int e = (int)(i >> 3), h = (int)(i & 7);
  int t = tgt[e];
  float mx = key2f(smax[(long)t * H_ + h]);
  float xd = dist[e] * (1.0f / CUTOFF_);
  float x2 = xd * xd, x4 = x2 * x2, x5 = x4 * xd, x6 = x5 * xd, x7 = x6 * xd;
  float env = (xd < 1.0f) ? (1.0f - 21.f * x5 + 35.f * x6 - 15.f * x7) : 0.f;
  float v = env * __expf(logits[i] - mx);
  ex[i] = v;
  atomicAdd(&den[(long)t * H_ + h], v);
}

// ------- value GEMM + gate + attention + scatter: agg[tgt] += (m@Wv)*g*attn
__global__ __launch_bounds__(128)
void k_v_scatter(const _Float16* m, const float* Wv, const _Float16* g,
                 const float* ex, const float* den, const int* tgt,
                 float* agg, int E) {
  __shared__ __align__(16) _Float16 sB[128 * 64];  // [n][k]
  __shared__ __align__(16) _Float16 sG[16 * 128];
  __shared__ __align__(16) _Float16 sA[16 * 64];
  __shared__ float sAt[16 * 8];
  __shared__ int sT[16];
  const int e0 = blockIdx.x * 16;
  const int tid = threadIdx.x, w = tid >> 5;
  if (tid < 16) sT[tid] = (e0 + tid < E) ? tgt[e0 + tid] : 0;
  for (int idx = tid; idx < 128 * 64; idx += 128) {
    int n = idx >> 6, k = idx & 63;
    sB[idx] = (_Float16)Wv[(long)k * C_ + n];
  }
  for (int idx = tid; idx < 256; idx += 128) {
    int r = idx >> 4, cc = (idx & 15) * 8;
    v8h v = {};
    if (e0 + r < E) v = *(const v8h*)(g + (long)(e0 + r) * C_ + cc);
    *(v8h*)(sG + r * C_ + cc) = v;
  }
  {
    int r = tid >> 3, h = tid & 7;
    int e = e0 + r;
    float a = 0.f;
    if (e < E) a = ex[(long)e * H_ + h] / (den[(long)tgt[e] * H_ + h] + 1e-16f);
    sAt[tid] = a;
  }
  __syncthreads();
  const int lane = tid & 31, nn = lane & 15, hi = lane >> 4;
  for (int kk = 0; kk < KM_; ++kk) {
    __syncthreads();
    for (int idx = tid; idx < 128; idx += 128) {
      int r = idx >> 3, cc = (idx & 7) * 8;
      v8h v = {};
      if (e0 + r < E) v = *(const v8h*)(m + (long)(e0 + r) * MROW_ + kk * HID_ + cc);
      *(v8h*)(sA + r * 64 + cc) = v;
    }
    __syncthreads();
#pragma unroll
    for (int cg = 0; cg < 2; ++cg) {
      v8f acc = {};
      acc = wmma32(fragA(sA, 64, 0),  fragB(sB, 64, cg * 64 + w * 16, 0),  acc);
      acc = wmma32(fragA(sA, 64, 32), fragB(sB, 64, cg * 64 + w * 16, 32), acc);
      int col = cg * 64 + w * 16 + nn;
      int h = col >> 4;
#pragma unroll
      for (int r = 0; r < 8; ++r) {
        int mm = r + 8 * hi, e = e0 + mm;
        if (e < E) {
          float v = acc[r] * (float)sG[mm * C_ + col] * sAt[mm * H_ + h];
          atomicAdd(&agg[(long)sT[mm] * (KM_ * C_) + kk * C_ + col], v);
        }
      }
    }
  }
}

// ------------- GEMM K=128 with epilogue modes; 64 rows x 64 cols per block.
// mode 0: xout[row*128+col] += acc            (ffn2 -> x, direct layout)
// mode 1: xout[n*XROW + M2F[k]*128+col] += acc (out-projection -> x)
// mode 2: outh[row*128+col] = f16(acc * gf[row/49][col])  (ffn1 gate)
__global__ __launch_bounds__(128)
void k_gemm_epi(const void* A, int a_f32, int lda, const float* B,
                float* xout, const _Float16* gf, _Float16* outh,
                int M, int mode) {
  __shared__ __align__(16) _Float16 sA[64 * 128];
  __shared__ __align__(16) _Float16 sB[64 * 32];
  const int r0 = blockIdx.x * 64, n0 = blockIdx.y * 64;
  const int tid = threadIdx.x, w = tid >> 5;
  if (a_f32) {
    const float* Af = (const float*)A;
    for (int idx = tid; idx < 8192; idx += 128) {
      int r = idx >> 7, k = idx & 127;
      sA[idx] = (_Float16)((r0 + r < M) ? Af[(long)(r0 + r) * lda + k] : 0.f);
    }
  } else {
    const _Float16* Ah = (const _Float16*)A;
    for (int idx = tid; idx < 1024; idx += 128) {
      int r = idx >> 4, cc = (idx & 15) * 8;
      v8h v = {};
      if (r0 + r < M) v = *(const v8h*)(Ah + (long)(r0 + r) * lda + cc);
      *(v8h*)(sA + r * 128 + cc) = v;
    }
  }
  const v8f vz = {};
  v8f acc[4] = {vz, vz, vz, vz};
  for (int ks = 0; ks < 4; ++ks) {
    __syncthreads();
    for (int idx = tid; idx < 2048; idx += 128) {
      int n = idx >> 5, k = idx & 31;
      sB[n * 32 + k] = (_Float16)B[(long)(ks * 32 + k) * C_ + n0 + n];
    }
    if (ks + 1 < 4)
      __builtin_prefetch((const void*)(B + (long)((ks + 1) * 32 + (tid >> 2)) * C_ +
                                      n0 + ((tid & 3) << 4)), 0, 1);
    __syncthreads();
    v16h bf = fragB(sB, 32, w * 16, 0);
#pragma unroll
    for (int rt = 0; rt < 4; ++rt)
      acc[rt] = wmma32(fragA(sA + rt * 16 * 128, 128, ks * 32), bf, acc[rt]);
  }
  const int lane = tid & 31, nn = lane & 15, hi = lane >> 4;
  const int col = n0 + w * 16 + nn;
#pragma unroll
  for (int rt = 0; rt < 4; ++rt) {
#pragma unroll
    for (int r = 0; r < 8; ++r) {
      int row = r0 + rt * 16 + r + 8 * hi;
      if (row < M) {
        float v = acc[rt][r];
        if (mode == 0) {
          xout[(long)row * C_ + col] += v;
        } else if (mode == 1) {
          int n = row / KM_, kk = row - n * KM_;
          xout[(long)n * XROW_ + c_M2F[kk] * C_ + col] += v;
        } else {
          int node = row / KF_;
          outh[(long)row * FFN_ + col] =
              (_Float16)(v * (float)gf[(long)node * FFN_ + col]);
        }
      }
    }
  }
}

__global__ void k_head_reduce(const _Float16* t, const float* Wh2, float* out, int N) {
  int n = blockIdx.x * blockDim.x + threadIdx.x;
  if (n >= N) return;
  float s = 0.f;
  for (int c = 0; c < C_; ++c) s += (float)t[(long)n * C_ + c] * Wh2[c];
  atomicAdd(out, s * (1.0f / AVG_NODES_));
}

// =========================================================================
extern "C" void kernel_launch(void* const* d_in, const int* in_sizes, int n_in,
                              void* d_out, int out_size, void* d_ws, size_t ws_size,
                              hipStream_t stream) {
  const int*   an    = (const int*)d_in[0];
  const int*   ei    = (const int*)d_in[1];
  const float* dist  = (const float*)d_in[2];
  const float* emb   = (const float*)d_in[3];
  const float* We0   = (const float*)d_in[4];
  const float* be0   = (const float*)d_in[5];
  const float* We1   = (const float*)d_in[6];
  const float* be1   = (const float*)d_in[7];
  const float* Weproj= (const float*)d_in[8];
  const float* gamma1= (const float*)d_in[9];
  const float* gamma2= (const float*)d_in[10];
  const float* Wr0   = (const float*)d_in[11];
  const float* Wr1   = (const float*)d_in[12];
  const float* Wrl   = (const float*)d_in[13];
  const float* Wmsg  = (const float*)d_in[14];
  const float* Wap   = (const float*)d_in[15];
  const float* wa    = (const float*)d_in[16];
  const float* Wg    = (const float*)d_in[17];
  const float* Wv    = (const float*)d_in[18];
  const float* Wo    = (const float*)d_in[19];
  const float* Wf1   = (const float*)d_in[20];
  const float* Wfg   = (const float*)d_in[21];
  const float* Wf2   = (const float*)d_in[22];
  const float* Wh1   = (const float*)d_in[23];
  const float* Wh2   = (const float*)d_in[24];
  (void)n_in; (void)ws_size;

  const int N = in_sizes[0];
  const int E = in_sizes[2];
  const int* src = ei;
  const int* tgt = ei + E;

  char* p = (char*)d_ws;
  auto alloc = [&](size_t bytes) -> char* {
    char* r = p;
    p += (bytes + 255) & ~(size_t)255;
    return r;
  };
  float*    x     = (float*)alloc((size_t)N * XROW_ * 4);
  _Float16* ym    = (_Float16*)alloc((size_t)N * KM_ * C_ * 2);
  _Float16* y2    = (_Float16*)alloc((size_t)N * KF_ * C_ * 2);
  _Float16* t0    = (_Float16*)alloc((size_t)E * C_ * 2);
  _Float16* hE    = (_Float16*)alloc((size_t)E * C_ * 2);
  _Float16* radf  = (_Float16*)alloc((size_t)E * C_ * 2);
  float*    radl  = (float*)alloc((size_t)E * 7 * 4);
  _Float16* mbuf  = (_Float16*)alloc((size_t)E * MROW_ * 2);
  _Float16* g     = (_Float16*)alloc((size_t)E * C_ * 2);
  float*    logits= (float*)alloc((size_t)E * H_ * 4);
  float*    exb   = (float*)alloc((size_t)E * H_ * 4);
  unsigned* smax  = (unsigned*)alloc((size_t)N * H_ * 4);
  float*    den   = (float*)alloc((size_t)N * H_ * 4);
  float*    agg   = (float*)alloc((size_t)N * KM_ * C_ * 4);
  _Float16* gf    = (_Float16*)alloc((size_t)N * C_ * 2);
  _Float16* hf    = (_Float16*)alloc((size_t)N * KF_ * C_ * 2);
  _Float16* th    = (_Float16*)alloc((size_t)N * C_ * 2);

  const int EB16 = (E + 15) / 16;
  const int EB64 = (E + 63) / 64;
  dim3 blk(128);

  k_init_x<<<N, 128, 0, stream>>>(x, emb, an, N);
  k_rbf_gemm<<<dim3(EB64, 2), blk, 0, stream>>>(dist, We0, be0, t0, E);
  k_gemm_silu<<<dim3(EB64, 2), blk, 0, stream>>>(t0, 0, C_, We1, C_, be1, hE, C_, E, C_);
  k_deg_scatter<<<dim3(EB64, 58), blk, 0, stream>>>(hE, Weproj, tgt, x, E);

  for (int i = 0; i < NL_; ++i) {
    k_rms<<<(unsigned)(N * KM_), 128, 0, stream>>>(x, gamma1 + i * C_, ym, N, KM_, 1);
    k_rbf_gemm<<<dim3(EB64, 2), blk, 0, stream>>>(dist, Wr0 + (size_t)i * NRBF_ * C_,
                                                  nullptr, t0, E);
    k_gemm_silu<<<dim3(EB64, 2), blk, 0, stream>>>(t0, 0, C_, Wr1 + (size_t)i * C_ * C_,
                                                   C_, nullptr, radf, C_, E, C_);
    k_radl<<<(E + 255) / 256, 256, 0, stream>>>(radf, Wrl + (size_t)i * C_ * 7, radl, E);
    k_msg<<<EB16, blk, 0, stream>>>(ym, src, tgt, Wmsg + (size_t)i * 2 * C_ * HID_,
                                    radl, mbuf, E);
    k_fill<<<512, 256, 0, stream>>>(logits, (long)E * H_, 0.f);
    k_ap<<<dim3(EB16, 4), blk, 0, stream>>>(mbuf, Wap + (size_t)i * HID_ * H_ * ALPHA_,
                                            wa + (size_t)i * H_ * ALPHA_, logits, E);
    k_fill<<<64, 256, 0, stream>>>((float*)smax, (long)N * H_, 0.f);
    k_segmax<<<(int)(((long)E * H_ + 255) / 256), 256, 0, stream>>>(logits, tgt, smax, E);
    k_fill<<<64, 256, 0, stream>>>(den, (long)N * H_, 0.f);
    k_exp_env<<<(int)(((long)E * H_ + 255) / 256), 256, 0, stream>>>(logits, smax, dist,
                                                                     tgt, exb, den, E);
    k_gemm_silu<<<dim3(EB64, 2), blk, 0, stream>>>(mbuf, 0, MROW_,
                                                   Wg + (size_t)i * HID_ * C_, C_,
                                                   nullptr, g, C_, E, HID_);
    k_fill<<<2048, 256, 0, stream>>>(agg, (long)N * KM_ * C_, 0.f);
    k_v_scatter<<<EB16, blk, 0, stream>>>(mbuf, Wv + (size_t)i * HID_ * C_, g, exb, den,
                                          tgt, agg, E);
    k_gemm_epi<<<dim3((N * KM_ + 63) / 64, 2), blk, 0, stream>>>(
        agg, 1, C_, Wo + (size_t)i * C_ * C_, x, nullptr, nullptr, N * KM_, 1);
    k_rms<<<(unsigned)(N * KF_), 128, 0, stream>>>(x, gamma2 + i * C_, y2, N, KF_, 0);
    k_gemm_silu<<<dim3((N + 63) / 64, 2), blk, 0, stream>>>(
        y2, 0, XROW_, Wfg + (size_t)i * C_ * C_, C_, nullptr, gf, C_, N, C_);
    k_gemm_epi<<<dim3((N * KF_ + 63) / 64, 2), blk, 0, stream>>>(
        y2, 0, C_, Wf1 + (size_t)i * C_ * C_, nullptr, gf, hf, N * KF_, 2);
    k_gemm_epi<<<dim3((N * KF_ + 63) / 64, 2), blk, 0, stream>>>(
        hf, 0, C_, Wf2 + (size_t)i * C_ * C_, x, nullptr, nullptr, N * KF_, 0);
  }

  k_gemm_silu<<<dim3((N + 63) / 64, 2), blk, 0, stream>>>(x, 1, XROW_, Wh1, C_,
                                                          nullptr, th, C_, N, C_);
  k_fill<<<1, 32, 0, stream>>>((float*)d_out, out_size, 0.f);
  k_head_reduce<<<(N + 255) / 256, 256, 0, stream>>>(th, Wh2, (float*)d_out, N);
}
